// SpectralConv_36197984371425
// MI455X (gfx1250) — matrix operand order repaired and checked by reading, not compiled, not run on hardware
//
#include <hip/hip_runtime.h>
#include <hip/hip_bf16.h>

typedef _Float16 half_t;
typedef __attribute__((ext_vector_type(16))) _Float16 v16h;
typedef __attribute__((ext_vector_type(8)))  _Float16 v8h;
typedef __attribute__((ext_vector_type(8)))  float    v8f;
typedef __attribute__((ext_vector_type(4)))  float    f32x4;

#define DEV static __device__ __forceinline__

// ---------------- problem constants ----------------
constexpr int B_ = 16, H_ = 256, W_ = 256, CI_ = 32, CO_ = 32, MX_ = 64;

// ---------------- workspace layout (half_t units) ----------------
constexpr size_t OFF_FWC  = 0;                       // [32ky][256w]  cos/16
constexpr size_t OFF_FWSN = OFF_FWC  + 32*256;       // -sin/16
constexpr size_t OFF_FHC  = OFF_FWSN + 32*256;       // [64m][256h]
constexpr size_t OFF_FHS  = OFF_FHC  + 64*256;
constexpr size_t OFF_FHSN = OFF_FHS  + 64*256;
constexpr size_t OFF_GHC  = OFF_FHSN + 64*256;       // [256h][64m]
constexpr size_t OFF_GHS  = OFF_GHC  + 256*64;
constexpr size_t OFF_GHSN = OFF_GHS  + 256*64;
constexpr size_t OFF_WIC  = OFF_GHSN + 256*64;       // [256w][32ky] with c_ky
constexpr size_t OFF_WISN = OFF_WIC  + 256*32;
constexpr size_t OFF_WRE  = OFF_WISN + 256*32;       // [64m][32ky][32co][32ci]
constexpr size_t NW       = (size_t)MX_*32*32*32;    // 2,097,152
constexpr size_t OFF_WIM  = OFF_WRE + NW;
constexpr size_t OFF_WIMN = OFF_WIM + NW;
constexpr size_t OFF_X1RE = OFF_WIMN + NW;           // [B][32ky][32ci][256h]
constexpr size_t NX1      = (size_t)B_*32*32*256;    // 4,194,304
constexpr size_t OFF_X1IM = OFF_X1RE + NX1;
constexpr size_t OFF_X2RE = OFF_X1IM + NX1;          // [32ky][64m][16b][32ci]
constexpr size_t NX2      = (size_t)32*64*16*32;     // 1,048,576
constexpr size_t OFF_X2IM = OFF_X2RE + NX2;
constexpr size_t OFF_YRE  = OFF_X2IM + NX2;          // [16b][32ky][32co][64m]
constexpr size_t OFF_YIM  = OFF_YRE + NX2;
constexpr size_t OFF_Y1RE = OFF_YIM + NX2;           // [B][256h][32co][32ky]
constexpr size_t OFF_Y1IM = OFF_Y1RE + NX1;
constexpr size_t WS_HALF  = OFF_Y1IM + NX1;          // ~55 MB

// ---------------- WMMA helper ----------------
DEV v8f wmma_f16(v16h a, v16h b, v8f c) {
  return __builtin_amdgcn_wmma_f32_16x16x32_f16(false, a, false, b, (short)0, c,
                                                false, false);
}

// ---------------- fragment loads (CDNA5 16-bit WMMA layouts) ----------------
// A 16x32, source row-major with row stride ld (halves), K contiguous.
// lane l -> row mBase+(l&15); lanes<16: K 8g..8g+7 and 16+8g..16+8g+7 (g=l>>4)
DEV v16h load_a_frag_g(const half_t* __restrict__ src, int ld, int mBase) {
  int l = threadIdx.x & 31;
  int row = mBase + (l & 15);
  int g = l >> 4;
  const half_t* p = src + (size_t)row * ld;
  v8h h0 = *(const v8h*)(p + 8 * g);
  v8h h1 = *(const v8h*)(p + 16 + 8 * g);
  v16h a;
#pragma unroll
  for (int i = 0; i < 8; ++i) { a[i] = h0[i]; a[8 + i] = h1[i]; }
  return a;
}
// B 32x16, source has K innermost: element (k, n) at src[n*colStride + k].
// lane l -> col nBase+(l&15); K 8g..8g+7 and 16+8g..16+8g+7 (g=l>>4)
DEV v16h load_b_frag_g(const half_t* __restrict__ src, int colStride, int nBase) {
  int l = threadIdx.x & 31;
  int col = nBase + (l & 15);
  int g = l >> 4;
  const half_t* p = src + (size_t)col * colStride;
  v8h h0 = *(const v8h*)(p + 8 * g);
  v8h h1 = *(const v8h*)(p + 16 + 8 * g);
  v16h b;
#pragma unroll
  for (int i = 0; i < 8; ++i) { b[i] = h0[i]; b[8 + i] = h1[i]; }
  return b;
}

// ---------------- C fragment stores (strided cols) ----------------
// C layout: VGPR r -> (M = mBase + r + 8*(lane/16), N = nBase + lane%16)
DEV void store_cfrag_f16(half_t* base, v8f acc, int rowStride, int colStride,
                         int mBase, int nBase) {
  int l = threadIdx.x & 31;
  int n = nBase + (l & 15);
  int r0 = mBase + ((l >> 4) << 3);
#pragma unroll
  for (int r = 0; r < 8; ++r)
    base[(size_t)(r0 + r) * rowStride + (size_t)n * colStride] = (half_t)acc[r];
}
DEV void store_cfrag_f32(float* base, v8f acc, int rowStride, int mBase, int nBase) {
  int l = threadIdx.x & 31;
  int n = nBase + (l & 15);
  int r0 = mBase + ((l >> 4) << 3);
#pragma unroll
  for (int r = 0; r < 8; ++r)
    base[(size_t)(r0 + r) * rowStride + n] = acc[r];
}

// ================= init kernels =================
__global__ void k_twiddles(half_t* ws) {
  int i = blockIdx.x * 256 + threadIdx.x;   // 131072 total
  const float S = 1.0f / 16.0f;
  const float TP = 6.283185307179586f / 256.0f;
  int j = i;
  if (j < 16384) {                 // Fw: [32ky][256w]
    int t = j >> 13; j &= 8191;
    int ky = j >> 8, w = j & 255;
    float a = TP * (float)((ky * w) & 255);
    ws[(t == 0 ? OFF_FWC : OFF_FWSN) + j] =
        (half_t)((t == 0 ? cosf(a) : -sinf(a)) * S);
    return;
  }
  j -= 16384;
  if (j < 49152) {                 // Fh: [64m][256h]
    int t = j / 16384; j %= 16384;
    int m = j >> 8, h = j & 255;
    int kx = (m < 32) ? m : m + 192;
    float a = TP * (float)((kx * h) & 255);
    float v = (t == 0) ? cosf(a) * S : (t == 1 ? sinf(a) * S : -sinf(a) * S);
    ws[(t == 0 ? OFF_FHC : (t == 1 ? OFF_FHS : OFF_FHSN)) + j] = (half_t)v;
    return;
  }
  j -= 49152;
  if (j < 49152) {                 // Gh: [256h][64m]
    int t = j / 16384; j %= 16384;
    int h = j >> 6, m = j & 63;
    int kx = (m < 32) ? m : m + 192;
    float a = TP * (float)((kx * h) & 255);
    float v = (t == 0) ? cosf(a) * S : (t == 1 ? sinf(a) * S : -sinf(a) * S);
    ws[(t == 0 ? OFF_GHC : (t == 1 ? OFF_GHS : OFF_GHSN)) + j] = (half_t)v;
    return;
  }
  j -= 49152;
  {                                 // Wi: [256w][32ky], hermitian factor c_ky
    int t = j / 8192; j %= 8192;
    int w = j >> 5, ky = j & 31;
    float cf = ((ky == 0) ? 1.0f : 2.0f) * S;
    float a = TP * (float)((ky * w) & 255);
    ws[(t == 0 ? OFF_WIC : OFF_WISN) + j] =
        (half_t)((t == 0 ? cosf(a) : -sinf(a)) * cf);
  }
}

// repack complex64 w[ci][co][m][ky] -> f16 planes [m64][ky][co][ci] (K=ci inner)
__global__ void k_weights(half_t* ws, const float* __restrict__ w1,
                          const float* __restrict__ w2) {
  int i = blockIdx.x * 256 + threadIdx.x;   // 2,097,152
  int ci = i & 31, co = (i >> 5) & 31, ky = (i >> 10) & 31, m = i >> 15;
  const float* src = (m < 32) ? w1 : w2;
  int mm = m & 31;
  size_t c = ((((size_t)ci * 32 + co) * 32) + mm) * 32 + ky;
  float re = src[2 * c], im = src[2 * c + 1];
  ws[OFF_WRE + i] = (half_t)re;
  ws[OFF_WIM + i] = (half_t)im;
  ws[OFF_WIMN + i] = (half_t)(-im);
}

// ================= stage 1: forward W-DFT =================
// per (b,h): C[32ky x 32ci] = Fw @ x[b,h].  Only stage needing LDS
// (f32->f16 convert + transpose of x so B frags are K(w)-contiguous).
__global__ void k_s1(const float* __restrict__ x, half_t* __restrict__ ws) {
  const half_t* FwC = ws + OFF_FWC;
  const half_t* FwSn = ws + OFF_FWSN;
  half_t* X1re = ws + OFF_X1RE;
  half_t* X1im = ws + OFF_X1IM;
  int bh = blockIdx.x, b = bh >> 8, h = bh & 255;
  __shared__ __align__(16) half_t lB[32 * 32];   // [ci][k] transposed
  int wid = threadIdx.x >> 5;
  int mB = (wid >> 1) * 16, nB = (wid & 1) * 16;
  v8f accRe = {0, 0, 0, 0, 0, 0, 0, 0}, accIm = {0, 0, 0, 0, 0, 0, 0, 0};
  const float* xb = x + (size_t)bh * (W_ * CI_);
  const int t = threadIdx.x;          // 128 threads: exactly one chunk each
  const int kk = t >> 2, cc = (t & 3) * 8;
  for (int kb = 0; kb < 8; ++kb) {
    __syncthreads();
    {  // straight-line staging: 32(w) x 32(ci) f32 tile -> LDS f16 transposed
      const float* src = xb + (size_t)kb * 32 * CI_;
      f32x4 v0 = *(const f32x4*)(src + kk * CI_ + cc);
      f32x4 v1 = *(const f32x4*)(src + kk * CI_ + cc + 4);
#pragma unroll
      for (int j = 0; j < 4; ++j) {
        lB[(cc + j) * 32 + kk]     = (half_t)v0[j];
        lB[(cc + 4 + j) * 32 + kk] = (half_t)v1[j];
      }
    }
    if (kb < 7)  // gfx1250 global_prefetch of next k-tile
      __builtin_prefetch(xb + (size_t)(kb + 1) * 32 * CI_ + t * 8, 0, 0);
    __syncthreads();
    v16h aC = load_a_frag_g(FwC + kb * 32, 256, mB);
    v16h aS = load_a_frag_g(FwSn + kb * 32, 256, mB);
    v16h bF = load_b_frag_g(lB, 32, nB);
    accRe = wmma_f16(aC, bF, accRe);
    accIm = wmma_f16(aS, bF, accIm);
  }
  // X1[b][ky][ci][h]: row=ky (stride 32*256), col=ci (stride 256)
  half_t* baseRe = X1re + (size_t)b * 262144 + h;
  half_t* baseIm = X1im + (size_t)b * 262144 + h;
  store_cfrag_f16(baseRe, accRe, 8192, 256, mB, nB);
  store_cfrag_f16(baseIm, accIm, 8192, 256, mB, nB);
}

// ================= stage 2: forward H-DFT (64 modes) =================
// per (b,ky,mt): C[32m x 32ci] complex, K=256(h).  No LDS: all frags global.
__global__ void k_s2(half_t* __restrict__ ws) {
  const half_t* FhC = ws + OFF_FHC;
  const half_t* FhS = ws + OFF_FHS;
  const half_t* FhSn = ws + OFF_FHSN;
  const half_t* X1re = ws + OFF_X1RE;
  const half_t* X1im = ws + OFF_X1IM;
  half_t* X2re = ws + OFF_X2RE;
  half_t* X2im = ws + OFF_X2IM;
  int mt = blockIdx.x & 1, ky = (blockIdx.x >> 1) & 31, b = blockIdx.x >> 6;
  int wid = threadIdx.x >> 5;
  int mB = (wid >> 1) * 16, nB = (wid & 1) * 16;
  v8f accRe = {0, 0, 0, 0, 0, 0, 0, 0}, accIm = {0, 0, 0, 0, 0, 0, 0, 0};
  const half_t* srcRe = X1re + (size_t)b * 262144 + (size_t)ky * 8192;
  const half_t* srcIm = X1im + (size_t)b * 262144 + (size_t)ky * 8192;
  const half_t* tabBase = FhC;  // FhC/FhS/FhSn share geometry
#pragma unroll
  for (int kb = 0; kb < 8; ++kb) {
    size_t tofs = (size_t)mt * 8192 + kb * 32;
    v16h aC  = load_a_frag_g(FhC + tofs, 256, mB);
    v16h aS  = load_a_frag_g(FhS + tofs, 256, mB);
    v16h aSn = load_a_frag_g(FhSn + tofs, 256, mB);
    v16h bRe = load_b_frag_g(srcRe + kb * 32, 256, nB);
    v16h bIm = load_b_frag_g(srcIm + kb * 32, 256, nB);
    accRe = wmma_f16(aC, bRe, accRe);   // cos@re
    accRe = wmma_f16(aS, bIm, accRe);   // + sin@im
    accIm = wmma_f16(aC, bIm, accIm);   // cos@im
    accIm = wmma_f16(aSn, bRe, accIm);  // - sin@re
  }
  (void)tabBase;
  // X2[ky][m][b][ci]: row=m (stride 16*32), col=ci (stride 1); b fixed
  half_t* baseRe = X2re + ((size_t)ky * 64 + mt * 32) * 512 + (size_t)b * 32;
  half_t* baseIm = X2im + ((size_t)ky * 64 + mt * 32) * 512 + (size_t)b * 32;
  store_cfrag_f16(baseRe, accRe, 512, 1, mB, nB);
  store_cfrag_f16(baseIm, accIm, 512, 1, mB, nB);
}

// ================= stage 3: per-mode channel mix =================
// per (m,ky): C[16b x 32co] complex = X2 @ W, K=32(ci). block=64, no LDS.
__global__ void k_s3(half_t* __restrict__ ws) {
  const half_t* X2re = ws + OFF_X2RE;
  const half_t* X2im = ws + OFF_X2IM;
  const half_t* WRe = ws + OFF_WRE;
  const half_t* WIm = ws + OFF_WIM;
  const half_t* WImn = ws + OFF_WIMN;
  half_t* Yre = ws + OFF_YRE;
  half_t* Yim = ws + OFF_YIM;
  int m = blockIdx.x >> 5, ky = blockIdx.x & 31;
  int nB = (threadIdx.x >> 5) * 16;
  const half_t* aBase = X2re + ((size_t)ky * 64 + m) * 512;  // [b][ci]
  const half_t* aBaseI = X2im + ((size_t)ky * 64 + m) * 512;
  const size_t wofs = ((size_t)m * 32 + ky) * 1024;          // [co][ci] tile
  v16h aRe = load_a_frag_g(aBase, 32, 0);
  v16h aIm = load_a_frag_g(aBaseI, 32, 0);
  v16h bRe = load_b_frag_g(WRe + wofs, 32, nB);
  v16h bIm = load_b_frag_g(WIm + wofs, 32, nB);
  v16h bImn = load_b_frag_g(WImn + wofs, 32, nB);
  v8f z = {0, 0, 0, 0, 0, 0, 0, 0};
  v8f accRe = wmma_f16(aRe, bRe, wmma_f16(aIm, bImn, z));  // re·Wre - im·Wim
  v8f accIm = wmma_f16(aRe, bIm, wmma_f16(aIm, bRe, z));   // re·Wim + im·Wre
  // Y[b][ky][co][m]: row=b (stride 32*32*64), col=co (stride 64)
  half_t* baseRe = Yre + (size_t)ky * 2048 + m;
  half_t* baseIm = Yim + (size_t)ky * 2048 + m;
  store_cfrag_f16(baseRe, accRe, 65536, 64, 0, nB);
  store_cfrag_f16(baseIm, accIm, 65536, 64, 0, nB);
}

// ================= stage 4: inverse H-DFT =================
// per (b,ky): B frags held in registers across all 8 h-tiles. K=64(m).
__global__ void k_s4(half_t* __restrict__ ws) {
  const half_t* GhC = ws + OFF_GHC;
  const half_t* GhS = ws + OFF_GHS;
  const half_t* GhSn = ws + OFF_GHSN;
  const half_t* Yre = ws + OFF_YRE;
  const half_t* Yim = ws + OFF_YIM;
  half_t* Y1re = ws + OFF_Y1RE;
  half_t* Y1im = ws + OFF_Y1IM;
  int ky = blockIdx.x & 31, b = blockIdx.x >> 5;
  int wid = threadIdx.x >> 5;
  int mB = (wid >> 1) * 16, nB = (wid & 1) * 16;
  const half_t* srcRe = Yre + (size_t)b * 65536 + (size_t)ky * 2048;
  const half_t* srcIm = Yim + (size_t)b * 65536 + (size_t)ky * 2048;
  // Y[b][ky][co][m]: col=co stride 64, K=m contiguous -> 4 resident B frags
  v16h bRe0 = load_b_frag_g(srcRe, 64, nB);
  v16h bRe1 = load_b_frag_g(srcRe + 32, 64, nB);
  v16h bIm0 = load_b_frag_g(srcIm, 64, nB);
  v16h bIm1 = load_b_frag_g(srcIm + 32, 64, nB);
#pragma unroll
  for (int ht = 0; ht < 8; ++ht) {
    v8f accRe = {0, 0, 0, 0, 0, 0, 0, 0}, accIm = {0, 0, 0, 0, 0, 0, 0, 0};
#pragma unroll
    for (int kb = 0; kb < 2; ++kb) {
      size_t tofs = (size_t)ht * 32 * 64 + kb * 32;
      v16h aC  = load_a_frag_g(GhC + tofs, 64, mB);
      v16h aS  = load_a_frag_g(GhS + tofs, 64, mB);
      v16h aSn = load_a_frag_g(GhSn + tofs, 64, mB);
      v16h bRe = kb ? bRe1 : bRe0;
      v16h bIm = kb ? bIm1 : bIm0;
      accRe = wmma_f16(aC, bRe, accRe);   // cos@re
      accRe = wmma_f16(aSn, bIm, accRe);  // - sin@im
      accIm = wmma_f16(aC, bIm, accIm);   // cos@im
      accIm = wmma_f16(aS, bRe, accIm);   // + sin@re
    }
    // Y1[b][h][co][ky]: row=h (stride 32*32), col=co (stride 32)
    half_t* baseRe = Y1re + ((size_t)b * 256 + ht * 32) * 1024 + ky;
    half_t* baseIm = Y1im + ((size_t)b * 256 + ht * 32) * 1024 + ky;
    store_cfrag_f16(baseRe, accRe, 1024, 32, mB, nB);
    store_cfrag_f16(baseIm, accIm, 1024, 32, mB, nB);
  }
}

// ================= stage 5: inverse rfft along W (real output) ===========
// per (b,h): B frags resident across all 8 w-tiles. K=32(ky).
__global__ void k_s5(const half_t* __restrict__ ws, float* __restrict__ out) {
  const half_t* WiC = ws + OFF_WIC;
  const half_t* WiSn = ws + OFF_WISN;
  const half_t* Y1re = ws + OFF_Y1RE;
  const half_t* Y1im = ws + OFF_Y1IM;
  const size_t bh = blockIdx.x;              // b*256 + h
  int wid = threadIdx.x >> 5;
  int mB = (wid >> 1) * 16, nB = (wid & 1) * 16;
  // Y1[b][h][co][ky]: col=co stride 32, K=ky contiguous
  v16h bRe = load_b_frag_g(Y1re + bh * 1024, 32, nB);
  v16h bIm = load_b_frag_g(Y1im + bh * 1024, 32, nB);
#pragma unroll
  for (int wt = 0; wt < 8; ++wt) {
    v16h aC  = load_a_frag_g(WiC + (size_t)wt * 32 * 32, 32, mB);
    v16h aSn = load_a_frag_g(WiSn + (size_t)wt * 32 * 32, 32, mB);
    v8f z = {0, 0, 0, 0, 0, 0, 0, 0};
    v8f acc = wmma_f16(aC, bRe, wmma_f16(aSn, bIm, z));
    float* base = out + (bh * 256 + (size_t)wt * 32) * 32;  // row=w, col=co
    store_cfrag_f32(base, acc, 32, mB, nB);
  }
}

// ================= launcher =================
extern "C" void kernel_launch(void* const* d_in, const int* in_sizes, int n_in,
                              void* d_out, int out_size, void* d_ws, size_t ws_size,
                              hipStream_t stream) {
  if (n_in < 3) return;
  const float* x  = (const float*)d_in[0];
  const float* w1 = (const float*)d_in[1];  // complex64 interleaved
  const float* w2 = (const float*)d_in[2];
  float* out = (float*)d_out;
  half_t* ws = (half_t*)d_ws;
  if (ws_size < WS_HALF * sizeof(half_t)) return;

  k_twiddles<<<512, 256, 0, stream>>>(ws);
  k_weights<<<8192, 256, 0, stream>>>(ws, w1, w2);
  k_s1<<<B_ * H_, 128, 0, stream>>>(x, ws);          // 4096 blocks
  k_s2<<<B_ * 32 * 2, 128, 0, stream>>>(ws);         // 1024 blocks
  k_s3<<<MX_ * 32, 64, 0, stream>>>(ws);             // 2048 blocks
  k_s4<<<B_ * 32, 128, 0, stream>>>(ws);             // 512 blocks
  k_s5<<<B_ * H_, 128, 0, stream>>>(ws, out);        // 4096 blocks
  (void)in_sizes; (void)out_size;
}
